// NewNewEncoder_42640435315105
// MI455X (gfx1250) — matrix operand order repaired and checked by reading, not compiled, not run on hardware
//
#include <hip/hip_runtime.h>
#include <math.h>

typedef __attribute__((ext_vector_type(16))) _Float16 v16h;
typedef __attribute__((ext_vector_type(8)))  _Float16 v8h;
typedef __attribute__((ext_vector_type(8)))  float    v8f;

// ---------------------------------------------------------------------------
// f32 -> f16 conversion (done once per layer)
// ---------------------------------------------------------------------------
__global__ void convert_f16_kernel(const float* __restrict__ src,
                                   _Float16* __restrict__ dst, long long count)
{
    long long i = (long long)blockIdx.x * blockDim.x + threadIdx.x;
    if (i >= count) return;
    dst[i] = (_Float16)src[i];
}

// W[K,64] f32 -> Wt[64,K] f16 (transposed so B fragments are contiguous in K)
__global__ void convert_wt_kernel(const float* __restrict__ W,
                                  _Float16* __restrict__ Wt, int K)
{
    int i = blockIdx.x * blockDim.x + threadIdx.x;
    if (i >= 64 * K) return;
    int n = i / K, k = i - n * K;
    Wt[i] = (_Float16)W[(size_t)k * 64 + n];
}

// ---------------------------------------------------------------------------
// WMMA GEMM: H[rows,64] = Xh[rows,K] @ W[K,64], f16 operands, f32 accum.
// One wave per 32-row strip: 2 A fragments x 4 column tiles = 8 acc tiles.
// Each B fragment load feeds two independent WMMAs (no D->A/B hazard).
// ---------------------------------------------------------------------------
template <int K>
__global__ void __launch_bounds__(256)
gat_gemm_wmma(const _Float16* __restrict__ Xh, const _Float16* __restrict__ WtH,
              float* __restrict__ H, int rows)
{
    const int lane = threadIdx.x & 31;
    const int wave = (blockIdx.x * blockDim.x + threadIdx.x) >> 5;
    const int row0 = wave * 32;
    const int mrow = lane & 15;        // A row / B,C column within tile
    const int half = lane >> 4;        // lane group 0/1

    int ar0 = row0 + mrow;       if (ar0 > rows - 1) ar0 = rows - 1; if (ar0 < 0) ar0 = 0;
    int ar1 = row0 + 16 + mrow;  if (ar1 > rows - 1) ar1 = rows - 1;

    v8f acc0[4] = {v8f{}, v8f{}, v8f{}, v8f{}};
    v8f acc1[4] = {v8f{}, v8f{}, v8f{}, v8f{}};

    const int kb = half * 16;          // A-frag K base (contiguous: K = kc + half*16 + e)
    const _Float16* xr0 = Xh + (size_t)ar0 * K;
    const _Float16* xr1 = Xh + (size_t)ar1 * K;

#pragma unroll
    for (int kc = 0; kc < K; kc += 32) {
        __builtin_prefetch(xr0 + kc + 32, 0, 3);

        // A fragments: elems 0..15 = K(kc+kb..+15), matching B
        v8h lo0 = *(const v8h*)(xr0 + kc + kb);
        v8h hi0 = *(const v8h*)(xr0 + kc + kb + 8);
        v16h a0 = __builtin_shufflevector(lo0, hi0, 0,1,2,3,4,5,6,7,8,9,10,11,12,13,14,15);
        v8h lo1 = *(const v8h*)(xr1 + kc + kb);
        v8h hi1 = *(const v8h*)(xr1 + kc + kb + 8);
        v16h a1 = __builtin_shufflevector(lo1, hi1, 0,1,2,3,4,5,6,7,8,9,10,11,12,13,14,15);

        // B fragment: col = n*16+mrow, K = kc + half*16 + e -> contiguous in WtH
#pragma unroll
        for (int n = 0; n < 4; ++n) {
            v16h b = *(const v16h*)(WtH + (size_t)(n * 16 + mrow) * K + kc + half * 16);
            acc0[n] = __builtin_amdgcn_wmma_f32_16x16x32_f16(
                false, a0, false, b, (short)0, acc0[n], false, false);
            acc1[n] = __builtin_amdgcn_wmma_f32_16x16x32_f16(
                false, a1, false, b, (short)0, acc1[n], false, false);
        }
    }
    // C/D layout: VGPR r -> M = r + 8*half, N = mrow
#pragma unroll
    for (int n = 0; n < 4; ++n) {
#pragma unroll
        for (int r = 0; r < 8; ++r) {
            int rowa = row0 + half * 8 + r;
            if (rowa < rows) H[(size_t)rowa * 64 + n * 16 + mrow] = acc0[n][r];
            int rowb = row0 + 16 + half * 8 + r;
            if (rowb < rows) H[(size_t)rowb * 64 + n * 16 + mrow] = acc1[n][r];
        }
    }
}

// generic runtime-K fallback (16 rows per wave)
__global__ void __launch_bounds__(256)
gat_gemm_wmma_gen(const _Float16* __restrict__ Xh, const _Float16* __restrict__ WtH,
                  float* __restrict__ H, int rows, int K)
{
    const int lane = threadIdx.x & 31;
    const int wave = (blockIdx.x * blockDim.x + threadIdx.x) >> 5;
    const int row0 = wave * 16;
    const int mrow = lane & 15;
    const int half = lane >> 4;

    int arow = row0 + mrow;
    if (arow > rows - 1) arow = rows - 1;
    if (arow < 0) arow = 0;

    v8f acc[4] = {v8f{}, v8f{}, v8f{}, v8f{}};
    const int kb = half * 16;
    const _Float16* xr = Xh + (size_t)arow * K;

    for (int kc = 0; kc < K; kc += 32) {
        v8h lo = *(const v8h*)(xr + kc + kb);
        v8h hi = *(const v8h*)(xr + kc + kb + 8);
        v16h a = __builtin_shufflevector(lo, hi, 0,1,2,3,4,5,6,7,8,9,10,11,12,13,14,15);
#pragma unroll
        for (int n = 0; n < 4; ++n) {
            v16h b = *(const v16h*)(WtH + (size_t)(n * 16 + mrow) * K + kc + half * 16);
            acc[n] = __builtin_amdgcn_wmma_f32_16x16x32_f16(
                false, a, false, b, (short)0, acc[n], false, false);
        }
    }
#pragma unroll
    for (int n = 0; n < 4; ++n)
#pragma unroll
        for (int r = 0; r < 8; ++r) {
            int row = row0 + half * 8 + r;
            if (row < rows) H[(size_t)row * 64 + n * 16 + mrow] = acc[n][r];
        }
}

// ---------------------------------------------------------------------------
// Per-node attention logits: as[i] = H[i].a_src ; ad[i] = H[i].a_dst
// ---------------------------------------------------------------------------
__global__ void alpha_kernel(const float* __restrict__ H,
                             const float* __restrict__ a_src,
                             const float* __restrict__ a_dst,
                             float* __restrict__ as_, float* __restrict__ ad_, int n)
{
    int i = blockIdx.x * blockDim.x + threadIdx.x;
    if (i >= n) return;
    const float4* h4 = (const float4*)(H + (size_t)i * 64);
    float s = 0.f, d = 0.f;
#pragma unroll
    for (int f = 0; f < 16; ++f) {
        float4 v = h4[f];
        s = fmaf(v.x, a_src[4*f+0], s); d = fmaf(v.x, a_dst[4*f+0], d);
        s = fmaf(v.y, a_src[4*f+1], s); d = fmaf(v.y, a_dst[4*f+1], d);
        s = fmaf(v.z, a_src[4*f+2], s); d = fmaf(v.z, a_dst[4*f+2], d);
        s = fmaf(v.w, a_src[4*f+3], s); d = fmaf(v.w, a_dst[4*f+3], d);
    }
    as_[i] = s; ad_[i] = d;
}

__global__ void init_nodes_kernel(float* __restrict__ m, float* __restrict__ den, int n)
{
    int i = blockIdx.x * blockDim.x + threadIdx.x;
    if (i >= n) return;
    m[i] = -__builtin_inff();
    den[i] = 0.f;
}

__global__ void init_out_kernel(float* __restrict__ out, const float* __restrict__ bias, int n64)
{
    int i = blockIdx.x * blockDim.x + threadIdx.x;
    if (i >= n64) return;
    out[i] = bias[i & 63];
}

__device__ __forceinline__ void atomic_max_f32(float* addr, float val)
{
    if (val >= 0.f) atomicMax((int*)addr, __float_as_int(val));
    else            atomicMin((unsigned int*)addr, __float_as_uint(val));
}

__device__ __forceinline__ void edge_endpoints(const long long* __restrict__ ei,
                                               int e, int E, int& s, int& d)
{
    if (e < E) { s = (int)ei[e]; d = (int)ei[(size_t)E + e]; }
    else       { s = d = e - E; }          // appended self-loop
}

__global__ void edge_max_kernel(const long long* __restrict__ ei,
                                const float* __restrict__ as_, const float* __restrict__ ad_,
                                float* __restrict__ m, int E, int n)
{
    int e = blockIdx.x * blockDim.x + threadIdx.x;
    if (e >= E + n) return;
    int s, d; edge_endpoints(ei, e, E, s, d);
    float v = as_[s] + ad_[d];
    v = v > 0.f ? v : 0.2f * v;            // LeakyReLU(0.2)
    atomic_max_f32(&m[d], v);
}

__global__ void edge_exp_kernel(const long long* __restrict__ ei,
                                const float* __restrict__ as_, const float* __restrict__ ad_,
                                const float* __restrict__ m,
                                float* __restrict__ wE, float* __restrict__ den, int E, int n)
{
    int e = blockIdx.x * blockDim.x + threadIdx.x;
    if (e >= E + n) return;
    int s, d; edge_endpoints(ei, e, E, s, d);
    float v = as_[s] + ad_[d];
    v = v > 0.f ? v : 0.2f * v;
    float ev = __expf(v - m[d]);
    wE[e] = ev;
    atomicAdd(&den[d], ev);
}

// normalize: wE[e] <- wE[e] / (den[dst]+eps)  (hoisted out of the 16x aggr pass)
__global__ void edge_alpha_kernel(const long long* __restrict__ ei,
                                  float* __restrict__ wE, const float* __restrict__ den,
                                  int E, int n)
{
    int e = blockIdx.x * blockDim.x + threadIdx.x;
    if (e >= E + n) return;
    int s, d; edge_endpoints(ei, e, E, s, d);
    wE[e] = wE[e] / (den[d] + 1e-16f);
}

// one thread per (edge, 4-feature quad): 16 threads per edge
__global__ void edge_aggr_kernel(const long long* __restrict__ ei,
                                 const float* __restrict__ wE,
                                 const float* __restrict__ H, float* __restrict__ out,
                                 int E, int n)
{
    long long t = (long long)blockIdx.x * blockDim.x + threadIdx.x;
    long long total = (long long)(E + n) * 16;
    if (t >= total) return;
    int e = (int)(t >> 4);
    int q = (int)(t & 15);
    int s, d; edge_endpoints(ei, e, E, s, d);
    float alpha = wE[e];
    const float4 hv = *(const float4*)(H + (size_t)s * 64 + q * 4);
    float* op = out + (size_t)d * 64 + q * 4;
    atomicAdd(op + 0, alpha * hv.x);
    atomicAdd(op + 1, alpha * hv.y);
    atomicAdd(op + 2, alpha * hv.z);
    atomicAdd(op + 3, alpha * hv.w);
}

// ---------------------------------------------------------------------------
static inline int ceil_div(long long a, long long b) { return (int)((a + b - 1) / b); }

static void run_gat_layer(const float* x, int K, const long long* ei,
                          const float* W, const float* a_src, const float* a_dst,
                          const float* bias, float* H, float* out,
                          float* as_, float* ad_, float* m, float* den, float* wE,
                          _Float16* Xh, _Float16* WtH,
                          int N, int E, hipStream_t stream)
{
    const int B = 256;
    long long nk = (long long)N * K;
    convert_f16_kernel<<<ceil_div(nk, B), B, 0, stream>>>(x, Xh, nk);
    convert_wt_kernel<<<ceil_div(64 * K, B), B, 0, stream>>>(W, WtH, K);

    if (K == 128) {
        int waves = ceil_div(N, 32);
        gat_gemm_wmma<128><<<ceil_div((long long)waves * 32, B), B, 0, stream>>>(Xh, WtH, H, N);
    } else if (K == 64) {
        int waves = ceil_div(N, 32);
        gat_gemm_wmma<64><<<ceil_div((long long)waves * 32, B), B, 0, stream>>>(Xh, WtH, H, N);
    } else {
        int waves = ceil_div(N, 16);
        gat_gemm_wmma_gen<<<ceil_div((long long)waves * 32, B), B, 0, stream>>>(Xh, WtH, H, N, K);
    }

    alpha_kernel<<<ceil_div(N, B), B, 0, stream>>>(H, a_src, a_dst, as_, ad_, N);
    init_nodes_kernel<<<ceil_div(N, B), B, 0, stream>>>(m, den, N);
    init_out_kernel<<<ceil_div((long long)N * 64, B), B, 0, stream>>>(out, bias, N * 64);
    int ET = E + N;
    edge_max_kernel<<<ceil_div(ET, B), B, 0, stream>>>(ei, as_, ad_, m, E, N);
    edge_exp_kernel<<<ceil_div(ET, B), B, 0, stream>>>(ei, as_, ad_, m, wE, den, E, N);
    edge_alpha_kernel<<<ceil_div(ET, B), B, 0, stream>>>(ei, wE, den, E, N);
    edge_aggr_kernel<<<ceil_div((long long)ET * 16, B), B, 0, stream>>>(ei, wE, H, out, E, N);
}

extern "C" void kernel_launch(void* const* d_in, const int* in_sizes, int n_in,
                              void* d_out, int out_size, void* d_ws, size_t ws_size,
                              hipStream_t stream)
{
    const float*     x    = (const float*)d_in[0];
    const long long* ei   = (const long long*)d_in[1];
    const float*     W1   = (const float*)d_in[2];
    const float*     asr1 = (const float*)d_in[3];
    const float*     adt1 = (const float*)d_in[4];
    const float*     b1   = (const float*)d_in[5];
    const float*     W2   = (const float*)d_in[6];
    const float*     asr2 = (const float*)d_in[7];
    const float*     adt2 = (const float*)d_in[8];
    const float*     b2   = (const float*)d_in[9];
    float*           out  = (float*)d_out;

    const int OUT = in_sizes[3];              // 64
    const int IN  = in_sizes[2] / OUT;        // 128
    const int N   = in_sizes[0] / IN;         // 50000
    const int E   = in_sizes[1] / 2;          // 800000
    (void)n_in; (void)out_size; (void)ws_size;

    // workspace layout
    float* ws   = (float*)d_ws;
    float* H    = ws;                          // N*64 f32
    float* out1 = H    + (size_t)N * 64;       // N*64 f32
    float* as_  = out1 + (size_t)N * 64;       // N
    float* ad_  = as_  + N;                    // N
    float* m    = ad_  + N;                    // N
    float* den  = m    + N;                    // N
    float* wE   = den  + N;                    // E+N
    _Float16* Xh  = (_Float16*)(wE + (E + N)); // N*IN f16 (reused at N*OUT for layer 2)
    _Float16* WtH = Xh + (size_t)N * IN;       // 64*IN f16

    // layer 1: x[N,IN] -> out1[N,64]
    run_gat_layer(x, IN, ei, W1, asr1, adt1, b1, H, out1,
                  as_, ad_, m, den, wE, Xh, WtH, N, E, stream);
    // layer 2: out1[N,64] -> out[N,64]
    run_gat_layer(out1, OUT, ei, W2, asr2, adt2, b2, H, out,
                  as_, ad_, m, den, wE, Xh, WtH, N, E, stream);
}
